// SequenceEncoder_66142496359003
// MI455X (gfx1250) — matrix operand order repaired and checked by reading, not compiled, not run on hardware
//
#include <hip/hip_runtime.h>
#include <hip/hip_bf16.h>
#include <math.h>

// ---------------------------------------------------------------------------
// Model dimensions (compile-time constants from the reference)
// ---------------------------------------------------------------------------
#define BATCH     2
#define TLEN      1024
#define IN_DIM    64
#define D_MODEL   512
#define NLAYERS   4
#define D_STATE   16
#define D_CONV    4
#define D_INNER   1024            // EXPAND * D_MODEL
#define DT_RANK   32
#define NTOK      (BATCH * TLEN)  // 2048 token rows
#define LN_EPS    1e-5f

// GEMM block tile: 8 waves arranged 4(M) x 2(N), each wave = 32x32 via 2x2 WMMA
#define BLK_M     128
#define BLK_N     64
#define BLK_K     32

typedef __attribute__((ext_vector_type(16))) __bf16 v16bf;
typedef __attribute__((ext_vector_type(8)))  __bf16 v8bf;
typedef __attribute__((ext_vector_type(8)))  float  v8f;
typedef __attribute__((ext_vector_type(4)))  int    v4i;

// ---------------------------------------------------------------------------
// Async global->LDS staging (gfx1250 GLOBAL_LOAD_ASYNC_TO_LDS_B128, ASYNCcnt).
// Builtin params (per hipcc diagnostic): (int4 as1*, int4 as3*, imm, imm).
// Falls back to regular load/store staging if the builtin is unavailable.
// ---------------------------------------------------------------------------
__device__ __forceinline__ void stage16(const __bf16* g, __bf16* l) {
#if defined(__gfx1250__) && __has_builtin(__builtin_amdgcn_global_load_async_to_lds_b128)
    __builtin_amdgcn_global_load_async_to_lds_b128(
        (__attribute__((address_space(1))) v4i*)(v4i*)(void*)const_cast<__bf16*>(g),
        (__attribute__((address_space(3))) v4i*)(v4i*)(void*)l,
        0, 0);
#else
    *(v8bf*)l = *(const v8bf*)g;
#endif
}
__device__ __forceinline__ void stage_wait() {
#if defined(__gfx1250__) && __has_builtin(__builtin_amdgcn_s_wait_asynccnt)
    __builtin_amdgcn_s_wait_asynccnt(0);
#endif
}

// ---------------------------------------------------------------------------
// f32 -> bf16 conversion
// ---------------------------------------------------------------------------
__global__ void cvt_f32_bf16_k(const float* __restrict__ src,
                               __bf16* __restrict__ dst, int n) {
    int i = blockIdx.x * blockDim.x + threadIdx.x;
    if (i < n) dst[i] = (__bf16)src[i];
}

// ---------------------------------------------------------------------------
// WMMA bf16 GEMM:  C[M,N] = A[M,K] * W[N,K]^T (+bias)
// A row-major (lda), W row-major (ldw), C row-major fp32 (ldc).
// Block = 256 threads = 8 waves -> 128x64 tile of C. Per K-step (32):
//   - async-stage A tile (128x32 bf16, 8KB) and B tile (64x32 bf16, 4KB) to LDS
//   - each wave builds fragments from LDS (ds_load_b128) and issues 4 WMMAs
//
// Fragment layouts follow CDNA5 ISA §7.12.2:
//  A (16x32 bf16): lane m=lane&15; half=lane>>4; elems 0..7 hold K=8*half..,
//                  elems 8..15 hold K=16+8*half..  -> two contiguous 16B loads
//  B (32x16 bf16): lane n=lane&15; half=lane>>4; elems 0..7 hold K=8*half..,
//                  elems 8..15 hold K=16+8*half..  -> two contiguous 16B loads
//  C/D (16x16 f32): lane n=lane&15; VGPR r -> m = 8*(lane>>4) + r
// ---------------------------------------------------------------------------
__device__ __forceinline__ v16bf frag_from_lds(const __bf16* p, int hiOff) {
    v8bf lo = *(const v8bf*)(p);
    v8bf hi = *(const v8bf*)(p + hiOff);
    v16bf r;
#pragma unroll
    for (int i = 0; i < 8; ++i) { r[i] = lo[i]; r[i + 8] = hi[i]; }
    return r;
}

__global__ void __launch_bounds__(256)
gemm_bf16_wmma_k(const __bf16* __restrict__ A, int lda,
                 const __bf16* __restrict__ W, int ldw,
                 float* __restrict__ C, int ldc,
                 const float* __restrict__ bias,
                 int K, int nBlocksN) {
    __shared__ __attribute__((aligned(64))) __bf16 Asm[BLK_M * BLK_K]; // [row][32]
    __shared__ __attribute__((aligned(64))) __bf16 Bsm[BLK_N * BLK_K]; // [row][32]

    const int tid  = threadIdx.x;
    const int bm   = blockIdx.x / nBlocksN;
    const int bn   = blockIdx.x - bm * nBlocksN;
    const int M0   = bm * BLK_M;
    const int N0   = bn * BLK_N;
    const int wave = tid >> 5;
    const int lane = tid & 31;
    const int wm   = (wave >> 1) * 32;      // wave M-offset in block tile
    const int wn   = (wave & 1) * 32;       // wave N-offset in block tile
    const int lm   = lane & 15;
    const int half = lane >> 4;

    v8f acc[2][2];
#pragma unroll
    for (int i = 0; i < 2; ++i)
#pragma unroll
        for (int j = 0; j < 2; ++j) acc[i][j] = (v8f)(0.0f);

    for (int k0 = 0; k0 < K; k0 += BLK_K) {
        __syncthreads();   // previous iteration's LDS reads complete
        // ---- stage A tile: 128 rows x 32 K = 512 x 16B chunks (2 per thread)
#pragma unroll
        for (int c = tid; c < BLK_M * 4; c += 256) {
            const int row = c >> 2, kk = (c & 3) * 8;
            stage16(A + (size_t)(M0 + row) * lda + k0 + kk, &Asm[row * BLK_K + kk]);
        }
        // ---- stage B tile: 64 rows x 32 K = 256 x 16B chunks (1 per thread)
#pragma unroll
        for (int c = tid; c < BLK_N * 4; c += 256) {
            const int row = c >> 2, kk = (c & 3) * 8;
            stage16(W + (size_t)(N0 + row) * ldw + k0 + kk, &Bsm[row * BLK_K + kk]);
        }
        // prefetch next K-slab into L2 while async copies land
        if (k0 + BLK_K < K)
            __builtin_prefetch(A + (size_t)(M0 + (tid >> 1)) * lda + k0 + BLK_K, 0, 3);
        stage_wait();
        __syncthreads();   // all waves' async copies visible

        v16bf af[2], bf[2];
#pragma unroll
        for (int i = 0; i < 2; ++i)
            af[i] = frag_from_lds(&Asm[(wm + 16 * i + lm) * BLK_K + 8 * half], 16);
#pragma unroll
        for (int j = 0; j < 2; ++j)
            bf[j] = frag_from_lds(&Bsm[(wn + 16 * j + lm) * BLK_K + 8 * half], 16);
#pragma unroll
        for (int i = 0; i < 2; ++i)
#pragma unroll
            for (int j = 0; j < 2; ++j)
                acc[i][j] = __builtin_amdgcn_wmma_f32_16x16x32_bf16(
                    false, af[i], false, bf[j], (short)0, acc[i][j], false, false);
    }

#pragma unroll
    for (int i = 0; i < 2; ++i)
#pragma unroll
        for (int j = 0; j < 2; ++j) {
            const int col = N0 + wn + 16 * j + lm;
            const float bv = bias ? bias[col] : 0.0f;
#pragma unroll
            for (int r = 0; r < 8; ++r) {
                const int row = M0 + wm + 16 * i + 8 * half + r;
                C[(size_t)row * ldc + col] = acc[i][j][r] + bv;
            }
        }
}

// ---------------------------------------------------------------------------
// Causal depthwise conv (D_CONV=4) + bias + SiLU. Reads xp from the packed
// xz buffer (stride 2*D_INNER), writes u (f32) and ub (bf16).
// ---------------------------------------------------------------------------
__global__ void conv_silu_k(const float* __restrict__ xz,
                            const float* __restrict__ cw,   // (D_INNER, 4)
                            const float* __restrict__ cb,   // (D_INNER)
                            float* __restrict__ u,
                            __bf16* __restrict__ ub) {
    int idx = blockIdx.x * blockDim.x + threadIdx.x;        // over NTOK*D_INNER
    if (idx >= NTOK * D_INNER) return;
    const int d  = idx & (D_INNER - 1);
    const int bt = idx >> 10;                               // token row
    const int t  = bt & (TLEN - 1);
    const int b  = bt >> 10;
    float acc = cb[d];
#pragma unroll
    for (int j = 0; j < D_CONV; ++j) {
        const int tt = t + j - (D_CONV - 1);
        if (tt >= 0)
            acc += xz[(size_t)(b * TLEN + tt) * (2 * D_INNER) + d] * cw[d * D_CONV + j];
    }
    const float s = acc / (1.0f + __expf(-acc));            // silu
    u[idx]  = s;
    ub[idx] = (__bf16)s;
}

// ---------------------------------------------------------------------------
// softplus in place: delta = log1p(exp(x)) (bias already added by GEMM)
// ---------------------------------------------------------------------------
__global__ void softplus_k(float* __restrict__ delta, int n) {
    int i = blockIdx.x * blockDim.x + threadIdx.x;
    if (i >= n) return;
    const float x = delta[i];
    delta[i] = (x > 20.0f) ? x : log1pf(__expf(x));
}

// ---------------------------------------------------------------------------
// Selective scan. 16 lanes = 16 states (n) of one channel (b,d); 2 channels
// per wave. Sequential over T; per-step y = sum_n(h_n * C_n) via shfl_xor
// reduction within the 16-lane group, + u*D, then SiLU(z) gating. Emits bf16
// y for the out_proj GEMM.
// ---------------------------------------------------------------------------
__global__ void __launch_bounds__(256)
scan_k(const float* __restrict__ delta,   // (NTOK, D_INNER)
       const float* __restrict__ u,       // (NTOK, D_INNER)
       const float* __restrict__ xdbl,    // (NTOK, 64): [dt(32) | B(16) | C(16)]
       const float* __restrict__ xz,      // (NTOK, 2*D_INNER): z at +D_INNER
       const float* __restrict__ A_log,   // (D_INNER, D_STATE)
       const float* __restrict__ Dp,      // (D_INNER)
       __bf16* __restrict__ yb) {         // (NTOK, D_INNER)
    const int lane = threadIdx.x & 31;
    const int gw   = blockIdx.x * (blockDim.x >> 5) + (threadIdx.x >> 5);
    const int ch   = gw * 2 + (lane >> 4);                  // 0..BATCH*D_INNER-1
    if (ch >= BATCH * D_INNER) return;
    const int n = lane & 15;
    const int b = ch >> 10;
    const int d = ch & (D_INNER - 1);

    const float A_dn = -__expf(A_log[d * D_STATE + n]);
    const float D_d  = Dp[d];
    float hn = 0.0f;

    for (int t = 0; t < TLEN; ++t) {
        const size_t bt = (size_t)(b * TLEN + t);
        const float dv = delta[bt * D_INNER + d];
        const float uv = u[bt * D_INNER + d];
        const float Bn = xdbl[bt * 64 + DT_RANK + n];
        const float Cn = xdbl[bt * 64 + DT_RANK + D_STATE + n];
        hn = __expf(dv * A_dn) * hn + dv * uv * Bn;
        float p = hn * Cn;
#pragma unroll
        for (int off = 8; off >= 1; off >>= 1)              // reduce within 16 lanes
            p += __shfl_xor(p, off, 32);
        if (n == 0) {
            const float yv = p + uv * D_d;
            const float zv = xz[bt * (2 * D_INNER) + D_INNER + d];
            const float yg = yv * (zv / (1.0f + __expf(-zv)));
            yb[bt * D_INNER + d] = (__bf16)yg;
        }
    }
}

// ---------------------------------------------------------------------------
// LayerNorm over 512 features, wave per row. res==nullptr -> plain LN.
// ---------------------------------------------------------------------------
__global__ void __launch_bounds__(256)
ln_k(const float* __restrict__ src, const float* __restrict__ res,
     float* __restrict__ dst, const float* __restrict__ g,
     const float* __restrict__ bta) {
    const int row  = blockIdx.x * (blockDim.x >> 5) + (threadIdx.x >> 5);
    if (row >= NTOK) return;
    const int lane = threadIdx.x & 31;
    const size_t base = (size_t)row * D_MODEL;

    float v[16];
    float sum = 0.0f;
#pragma unroll
    for (int i = 0; i < 16; ++i) {
        const int e = i * 32 + lane;
        float x = src[base + e];
        if (res) x += res[base + e];
        v[i] = x;
        sum += x;
    }
#pragma unroll
    for (int off = 16; off >= 1; off >>= 1) sum += __shfl_xor(sum, off, 32);
    const float mean = sum * (1.0f / D_MODEL);

    float vs = 0.0f;
#pragma unroll
    for (int i = 0; i < 16; ++i) { const float dd = v[i] - mean; vs += dd * dd; }
#pragma unroll
    for (int off = 16; off >= 1; off >>= 1) vs += __shfl_xor(vs, off, 32);
    const float rstd = rsqrtf(vs * (1.0f / D_MODEL) + LN_EPS);

#pragma unroll
    for (int i = 0; i < 16; ++i) {
        const int e = i * 32 + lane;
        dst[base + e] = (v[i] - mean) * rstd * g[e] + bta[e];
    }
}

// ---------------------------------------------------------------------------
// Mean-pool over T: pooled[b,c] = mean_t seq[b,t,c]
// ---------------------------------------------------------------------------
__global__ void pool_k(const float* __restrict__ seq, float* __restrict__ pooled) {
    int i = blockIdx.x * blockDim.x + threadIdx.x;          // 0..BATCH*D_MODEL-1
    if (i >= BATCH * D_MODEL) return;
    const int b = i >> 9, c = i & (D_MODEL - 1);
    float s = 0.0f;
    for (int t = 0; t < TLEN; ++t)
        s += seq[(size_t)(b * TLEN + t) * D_MODEL + c];
    pooled[i] = s * (1.0f / TLEN);
}

// ---------------------------------------------------------------------------
// Host-side helpers
// ---------------------------------------------------------------------------
static inline void launch_cvt(const float* s, __bf16* d, int n, hipStream_t st) {
    cvt_f32_bf16_k<<<(n + 255) / 256, 256, 0, st>>>(s, d, n);
}
static inline void launch_gemm(const __bf16* A, int lda, const __bf16* W, int ldw,
                               float* C, int ldc, const float* bias,
                               int M, int N, int K, hipStream_t st) {
    const int nbn    = N / BLK_N;
    const int blocks = (M / BLK_M) * nbn;
    gemm_bf16_wmma_k<<<blocks, 256, 0, st>>>(A, lda, W, ldw, C, ldc, bias, K, nbn);
}

extern "C" void kernel_launch(void* const* d_in, const int* in_sizes, int n_in,
                              void* d_out, int out_size, void* d_ws, size_t ws_size,
                              hipStream_t stream) {
    (void)in_sizes; (void)n_in; (void)out_size; (void)ws_size;
    // ---- inputs (setup_inputs order) ----
    const float* x       = (const float*)d_in[0];   // (2,1024,64)
    const float* ipw     = (const float*)d_in[1];   // (512,64)
    const float* ipb     = (const float*)d_in[2];   // (512)
    const float* inpw    = (const float*)d_in[3];   // (4,2048,512)
    const float* convw   = (const float*)d_in[4];   // (4,1024,1,4)
    const float* convb   = (const float*)d_in[5];   // (4,1024)
    const float* xpw     = (const float*)d_in[6];   // (4,64,1024)
    const float* dtw     = (const float*)d_in[7];   // (4,1024,32)
    const float* dtb     = (const float*)d_in[8];   // (4,1024)
    const float* A_log   = (const float*)d_in[9];   // (4,1024,16)
    const float* Dp      = (const float*)d_in[10];  // (4,1024)
    const float* outw    = (const float*)d_in[11];  // (4,512,1024)
    const float* lng     = (const float*)d_in[12];  // (4,512)
    const float* lnb     = (const float*)d_in[13];  // (4,512)
    const float* fing    = (const float*)d_in[14];  // (512)
    const float* finb    = (const float*)d_in[15];  // (512)

    // ---- workspace bump allocator ----
    char* p = (char*)d_ws;
    auto alloc = [&](size_t bytes) -> void* {
        void* r = (void*)p; p += (bytes + 255) & ~(size_t)255; return r;
    };
    float*  h      = (float*) alloc((size_t)NTOK * D_MODEL * 4);
    float*  xz     = (float*) alloc((size_t)NTOK * 2 * D_INNER * 4);
    float*  u      = (float*) alloc((size_t)NTOK * D_INNER * 4);
    float*  xdbl   = (float*) alloc((size_t)NTOK * 64 * 4);
    float*  delta  = (float*) alloc((size_t)NTOK * D_INNER * 4);
    float*  outb   = (float*) alloc((size_t)NTOK * D_MODEL * 4);
    __bf16* xb     = (__bf16*)alloc((size_t)NTOK * IN_DIM * 2);
    __bf16* hb     = (__bf16*)alloc((size_t)NTOK * D_MODEL * 2);
    __bf16* ub     = (__bf16*)alloc((size_t)NTOK * D_INNER * 2);
    __bf16* xdblb  = (__bf16*)alloc((size_t)NTOK * 64 * 2);
    __bf16* yb     = (__bf16*)alloc((size_t)NTOK * D_INNER * 2);
    __bf16* ipwb   = (__bf16*)alloc((size_t)D_MODEL * IN_DIM * 2);
    __bf16* inpwb  = (__bf16*)alloc((size_t)NLAYERS * 2 * D_INNER * D_MODEL * 2);
    __bf16* xpwb   = (__bf16*)alloc((size_t)NLAYERS * 64 * D_INNER * 2);
    __bf16* dtwb   = (__bf16*)alloc((size_t)NLAYERS * D_INNER * DT_RANK * 2);
    __bf16* outwb  = (__bf16*)alloc((size_t)NLAYERS * D_MODEL * D_INNER * 2);

    // ---- weight & input conversions (deterministic each call) ----
    launch_cvt(x,    xb,    NTOK * IN_DIM,                       stream);
    launch_cvt(ipw,  ipwb,  D_MODEL * IN_DIM,                    stream);
    launch_cvt(inpw, inpwb, NLAYERS * 2 * D_INNER * D_MODEL,     stream);
    launch_cvt(xpw,  xpwb,  NLAYERS * 64 * D_INNER,              stream);
    launch_cvt(dtw,  dtwb,  NLAYERS * D_INNER * DT_RANK,         stream);
    launch_cvt(outw, outwb, NLAYERS * D_MODEL * D_INNER,         stream);

    // ---- input projection: h = x @ ipw^T + ipb ----
    launch_gemm(xb, IN_DIM, ipwb, IN_DIM, h, D_MODEL, ipb,
                NTOK, D_MODEL, IN_DIM, stream);

    const int elemsDI = NTOK * D_INNER;
    for (int l = 0; l < NLAYERS; ++l) {
        // bf16 copy of residual stream
        launch_cvt(h, hb, NTOK * D_MODEL, stream);
        // xz = h @ in_proj^T    (M=2048, N=2048, K=512)
        launch_gemm(hb, D_MODEL, inpwb + (size_t)l * 2 * D_INNER * D_MODEL, D_MODEL,
                    xz, 2 * D_INNER, nullptr, NTOK, 2 * D_INNER, D_MODEL, stream);
        // depthwise causal conv + SiLU -> u, ub
        conv_silu_k<<<(elemsDI + 255) / 256, 256, 0, stream>>>(
            xz, convw + (size_t)l * D_INNER * D_CONV, convb + (size_t)l * D_INNER,
            u, ub);
        // x_dbl = u @ x_proj^T  (M=2048, N=64, K=1024)
        launch_gemm(ub, D_INNER, xpwb + (size_t)l * 64 * D_INNER, D_INNER,
                    xdbl, 64, nullptr, NTOK, 64, D_INNER, stream);
        launch_cvt(xdbl, xdblb, NTOK * 64, stream);
        // delta_raw = dt @ dt_proj^T + dt_b  (M=2048, N=1024, K=32; lda=64)
        launch_gemm(xdblb, 64, dtwb + (size_t)l * D_INNER * DT_RANK, DT_RANK,
                    delta, D_INNER, dtb + (size_t)l * D_INNER,
                    NTOK, D_INNER, DT_RANK, stream);
        softplus_k<<<(elemsDI + 255) / 256, 256, 0, stream>>>(delta, elemsDI);
        // selective scan + SiLU(z) gating -> yb (bf16)
        scan_k<<<(BATCH * D_INNER / 2 + 7) / 8, 256, 0, stream>>>(
            delta, u, xdbl, xz,
            A_log + (size_t)l * D_INNER * D_STATE, Dp + (size_t)l * D_INNER, yb);
        // out = y @ out_proj^T  (M=2048, N=512, K=1024)
        launch_gemm(yb, D_INNER, outwb + (size_t)l * D_MODEL * D_INNER, D_INNER,
                    outb, D_MODEL, nullptr, NTOK, D_MODEL, D_INNER, stream);
        // h = LN(out + h)
        ln_k<<<(NTOK + 7) / 8, 256, 0, stream>>>(
            outb, h, h, lng + (size_t)l * D_MODEL, lnb + (size_t)l * D_MODEL);
    }

    // ---- final LN -> seq_latent, then mean pool ----
    float* seq    = (float*)d_out;
    float* pooled = seq + (size_t)NTOK * D_MODEL;
    ln_k<<<(NTOK + 7) / 8, 256, 0, stream>>>(h, nullptr, seq, fing, finb);
    pool_k<<<(BATCH * D_MODEL + 255) / 256, 256, 0, stream>>>(seq, pooled);
}
